// MultiHeadAttention_60138132078900
// MI455X (gfx1250) — compile-verified, matrix operand order checked by reading, not run on hardware
//
#include <hip/hip_runtime.h>
#include <hip/hip_bf16.h>

typedef __bf16 bf16;
typedef __attribute__((ext_vector_type(16))) __bf16 v16bf;
typedef __attribute__((ext_vector_type(8)))  __bf16 v8bf;
typedef __attribute__((ext_vector_type(8)))  float  v8f;
typedef __attribute__((ext_vector_type(4)))  unsigned int v4u;
typedef __attribute__((ext_vector_type(8)))  int v8i;
typedef __attribute__((ext_vector_type(4)))  int v4i;

#define T_LEN 2048
#define BATCH 2
#define EMB   1024
#define HEADS 16
#define HD    64
#define MROWS (T_LEN * BATCH)   // 4096

#if defined(__has_builtin)
#if __has_builtin(__builtin_amdgcn_tensor_load_to_lds) && __has_builtin(__builtin_amdgcn_s_wait_tensorcnt)
#define USE_TDM 1
#endif
#endif
#ifndef USE_TDM
#define USE_TDM 0
#endif

// --- fragment helpers -------------------------------------------------------
// K-index pattern for bf16 A/B fragments (16x32): lanes 0-15 hold K in
// {0..7, 16..23}, lanes 16-31 hold K in {8..15, 24..31}.
__device__ __forceinline__ int kpat(int e, int kh) {
  return (e < 8) ? (e + kh * 8) : (e + 8 + kh * 8);
}

// Load a 16x32 A-type (or K-contiguous B-type) fragment: p = row start + k0.
// Two 16-byte contiguous loads per lane (global or LDS).
__device__ __forceinline__ v16bf load_frag_contig(const bf16* p, int kh) {
  union { v16bf v; v8bf h[2]; } u;
  u.h[0] = *(const v8bf*)(p + kh * 8);
  u.h[1] = *(const v8bf*)(p + 16 + kh * 8);
  return u.v;
}

__device__ __forceinline__ v8f wmma_bf16(v16bf a, v16bf b, v8f c) {
  return __builtin_amdgcn_wmma_f32_16x16x32_bf16(false, a, false, b,
                                                 (short)0, c, false, false);
}

#if USE_TDM
// Issue one TDM load of a 64x64 bf16 tile (row stride BATCH*EMB elements) into
// LDS at byte offset lds_addr. D# packing per CDNA5 ISA ch.8 (2-D tensor,
// groups 2/3 zero). Wave-level op, tracked by TENSORcnt.
__device__ __forceinline__ void tdm_load_tile64(unsigned lds_addr, const bf16* gptr) {
  const unsigned long long ga = (unsigned long long)(uintptr_t)gptr;
  v4u g0 = {};
  g0[0] = 1u;                                            // count=1, user mode
  g0[1] = lds_addr;                                      // LDS byte address
  g0[2] = (unsigned)(ga & 0xFFFFFFFFu);                  // global_addr[31:0]
  g0[3] = (unsigned)((ga >> 32) & 0x01FFFFFFu) | (2u << 30);  // [56:32] | type=2
  const unsigned td0 = 64, td1 = 64, tile0 = 64, tile1 = 64;
  const unsigned long long st0 = (unsigned long long)BATCH * EMB;  // elems
  v8i g1 = {};
  g1[0] = (int)(1u << 16);                               // data_size = 2 bytes
  g1[1] = (int)((td0 & 0xFFFFu) << 16);                  // tensor_dim0 lo16
  g1[2] = (int)(((td0 >> 16) & 0xFFFFu) | ((td1 & 0xFFFFu) << 16));
  g1[3] = (int)(((td1 >> 16) & 0xFFFFu) | (tile0 << 16));
  g1[4] = (int)(tile1 & 0xFFFFu);                        // tile_dim1, tile_dim2=0
  g1[5] = (int)(st0 & 0xFFFFFFFFu);                      // dim0 stride lo32
  g1[6] = (int)((st0 >> 32) & 0xFFFFu);                  // dim0 stride hi16
  g1[7] = 0;
  v4i z4 = {};
#if defined(__clang_major__) && (__clang_major__ >= 23)
  v8i z8 = {};
  __builtin_amdgcn_tensor_load_to_lds(g0, g1, z4, z4, z8, 0);
#else
  __builtin_amdgcn_tensor_load_to_lds(g0, g1, z4, z4, 0);
#endif
}
#endif

// --- fp32 -> bf16 conversion (8 elements / thread / iter) -------------------
__global__ __launch_bounds__(256)
void f32_to_bf16_kernel(const float* __restrict__ in, bf16* __restrict__ out, int n8) {
  int i = blockIdx.x * blockDim.x + threadIdx.x;
  const int stride = gridDim.x * blockDim.x;
  for (; i < n8; i += stride) {
    const float4 a = ((const float4*)in)[2 * i];
    const float4 b = ((const float4*)in)[2 * i + 1];
    union { v8bf v; bf16 e[8]; } u;
    u.e[0] = (bf16)a.x; u.e[1] = (bf16)a.y; u.e[2] = (bf16)a.z; u.e[3] = (bf16)a.w;
    u.e[4] = (bf16)b.x; u.e[5] = (bf16)b.y; u.e[6] = (bf16)b.z; u.e[7] = (bf16)b.w;
    ((v8bf*)out)[i] = u.v;
  }
}

// --- GEMM: P[m,n] = sum_k X[m,k] * W[n,k] + bias[n] -------------------------
// X row-major [M,K] bf16, W row-major [N,K] bf16. One wave -> 16x128 strip.
template <bool OUT_F32>
__global__ __launch_bounds__(256)
void gemm_xwt_bias(const bf16* __restrict__ X, const bf16* __restrict__ W,
                   const float* __restrict__ bias, void* __restrict__ out,
                   int M, int N, int K) {
  const int lane = threadIdx.x & 31;
  const int wv   = threadIdx.x >> 5;
  const int kh   = lane >> 4;
  const int ln   = lane & 15;
  const int m0   = blockIdx.y * 128 + wv * 16;
  const int n0   = blockIdx.x * 128;

  const bf16* arow = X + (size_t)(m0 + ln) * K;
  const bf16* brow = W + (size_t)(n0 + ln) * K;

  v8f acc[8] = {};
#pragma unroll 2
  for (int kk = 0; kk < K; kk += 32) {
    __builtin_prefetch(arow + kk + 128, 0, 1);
    v16bf a = load_frag_contig(arow + kk, kh);
#pragma unroll
    for (int j = 0; j < 8; ++j) {
      v16bf b = load_frag_contig(brow + (size_t)(16 * j) * K + kk, kh);
      acc[j] = wmma_bf16(a, b, acc[j]);
    }
  }

#pragma unroll
  for (int j = 0; j < 8; ++j) {
    const int col = n0 + ln + 16 * j;
    const float bv = bias[col];
#pragma unroll
    for (int e = 0; e < 8; ++e) {
      const int row = m0 + e + kh * 8;
      const float v = acc[j][e] + bv;
      if (OUT_F32) ((float*)out)[(size_t)row * N + col] = v;
      else         ((bf16*)out)[(size_t)row * N + col]  = (bf16)v;
    }
  }
}

// --- flash attention --------------------------------------------------------
// One block = 8 waves = one (b, h, 128-row Q panel); each wave owns a 16-row
// tile. K/V 64x64 tiles are double-buffered in LDS; with TDM the next step's
// tiles are DMA'd by the Tensor Data Mover while the current step computes
// (sync via TENSORcnt: wait<=2 keeps one tile-pair in flight, in-order
// completion guarantees the current pair is resident). P is bounced through
// wave-private LDS (same-wave DS ops are in-order) to regain A-frag layout.
__global__ __launch_bounds__(256)
void flash_attn_kernel(const bf16* __restrict__ Q, const bf16* __restrict__ Kp,
                       const bf16* __restrict__ Vp, bf16* __restrict__ CTX,
                       float* __restrict__ Mst, float* __restrict__ Lst) {
  __shared__ __align__(16) bf16 Kt[2][64 * 64];   // 16 KB
  __shared__ __align__(16) bf16 Vt[2][64 * 64];   // 16 KB
  __shared__ __align__(16) bf16 Pl[8][16 * 64];   // 16 KB, wave-private slices

  const int tid  = threadIdx.x;
  const int lane = tid & 31;
  const int wv   = tid >> 5;
  const int kh   = lane >> 4;
  const int ln   = lane & 15;

  const int TB = T_LEN / 128;                 // 16 panels per (b,h)
  const int b   = blockIdx.x / (HEADS * TB);
  const int r   = blockIdx.x % (HEADS * TB);
  const int h   = r / TB;
  const int t0  = (r % TB) * 128 + wv * 16;
  const int hoff = h * HD;
  const float scale = 0.125f;                 // 1/sqrt(64)

  const bf16* qrow = Q + ((size_t)(t0 + ln) * BATCH + b) * EMB + hoff;
  const v16bf qf0 = load_frag_contig(qrow + 0,  kh);
  const v16bf qf1 = load_frag_contig(qrow + 32, kh);

  float mrow[8], lrow[8];
#pragma unroll
  for (int e = 0; e < 8; ++e) { mrow[e] = -1e30f; lrow[e] = 0.0f; }
  v8f o[4] = {};
  bf16* pl = Pl[wv];

#if USE_TDM
  // Prologue: wave 0 kicks off the first K/V tile pair.
  if (wv == 0) {
    tdm_load_tile64((unsigned)(uintptr_t)&Kt[0][0],
                    Kp + ((size_t)0 * BATCH + b) * EMB + hoff);
    tdm_load_tile64((unsigned)(uintptr_t)&Vt[0][0],
                    Vp + ((size_t)0 * BATCH + b) * EMB + hoff);
  }
#endif

  for (int s0 = 0; s0 < T_LEN; s0 += 64) {
    const int buf = (s0 >> 6) & 1;
    __syncthreads();  // all reads of the buffer about to be overwritten done
#if USE_TDM
    if (wv == 0) {
      if (s0 + 64 < T_LEN) {
        const size_t g = ((size_t)(s0 + 64) * BATCH + b) * EMB + hoff;
        tdm_load_tile64((unsigned)(uintptr_t)&Kt[buf ^ 1][0], Kp + g);
        tdm_load_tile64((unsigned)(uintptr_t)&Vt[buf ^ 1][0], Vp + g);
        __builtin_amdgcn_s_wait_tensorcnt(2);  // pair for this step resident
      } else {
        __builtin_amdgcn_s_wait_tensorcnt(0);
      }
    }
#else
    // Cooperative staging fallback: 512 x 16B chunks each, 2 per thread.
#pragma unroll
    for (int c = tid; c < 512; c += 256) {
      const int row  = c >> 3;
      const int col8 = (c & 7) * 8;
      const size_t g = ((size_t)(s0 + row) * BATCH + b) * EMB + hoff + col8;
      *(v8bf*)(&Kt[buf][0] + row * 64 + col8) = *(const v8bf*)(Kp + g);
      *(v8bf*)(&Vt[buf][0] + row * 64 + col8) = *(const v8bf*)(Vp + g);
    }
#endif
    __syncthreads();
    const bf16* Ktile = &Kt[buf][0];
    const bf16* Vtile = &Vt[buf][0];

    // ---- scores: four 16x16 tiles over 64 S-columns (8 WMMA) ----
    v8f sc[4];
#pragma unroll
    for (int st = 0; st < 4; ++st) {
      const bf16* kr = Ktile + (st * 16 + ln) * 64;
      v8f s_ = {};
      s_ = wmma_bf16(qf0, load_frag_contig(kr + 0,  kh), s_);
      s_ = wmma_bf16(qf1, load_frag_contig(kr + 32, kh), s_);
      sc[st] = s_;
    }

    // ---- online softmax over 64 columns (one reduction set per step) ----
    float p[4][8], rsc[8];
#pragma unroll
    for (int e = 0; e < 8; ++e) {
      float a0 = sc[0][e] * scale, a1 = sc[1][e] * scale;
      float a2 = sc[2][e] * scale, a3 = sc[3][e] * scale;
      float mx = fmaxf(fmaxf(a0, a1), fmaxf(a2, a3));
#pragma unroll
      for (int off = 1; off < 16; off <<= 1) mx = fmaxf(mx, __shfl_xor(mx, off, 32));
      const float mnew = fmaxf(mrow[e], mx);
      const float s    = __expf(mrow[e] - mnew);
      const float e0 = __expf(a0 - mnew), e1 = __expf(a1 - mnew);
      const float e2 = __expf(a2 - mnew), e3 = __expf(a3 - mnew);
      float rs = (e0 + e1) + (e2 + e3);
#pragma unroll
      for (int off = 1; off < 16; off <<= 1) rs += __shfl_xor(rs, off, 32);
      lrow[e] = lrow[e] * s + rs;
      mrow[e] = mnew;
      p[0][e] = e0; p[1][e] = e1; p[2][e] = e2; p[3][e] = e3;
      rsc[e] = s;
    }
#pragma unroll
    for (int j = 0; j < 4; ++j)
#pragma unroll
      for (int e = 0; e < 8; ++e) o[j][e] *= rsc[e];

    // ---- P (C-layout) -> wave-private LDS -> A-fragment layout ----
#pragma unroll
    for (int e = 0; e < 8; ++e)
#pragma unroll
      for (int st = 0; st < 4; ++st)
        pl[(e + kh * 8) * 64 + st * 16 + ln] = (bf16)p[st][e];
    asm volatile("" ::: "memory");  // same-wave LDS ops are in-order; stop
                                    // compiler reordering only.
    const v16bf pa0 = load_frag_contig(pl + ln * 64 + 0,  kh);
    const v16bf pa1 = load_frag_contig(pl + ln * 64 + 32, kh);

    // ---- P @ V : 8 WMMA over D = 64 (K = 64 split in two) ----
#pragma unroll
    for (int j = 0; j < 4; ++j) {
      union { v16bf v; bf16 e[16]; } vf0, vf1;
#pragma unroll
      for (int e = 0; e < 16; ++e) {
        vf0.e[e] = Vtile[(kpat(e, kh)) * 64      + j * 16 + ln];
        vf1.e[e] = Vtile[(32 + kpat(e, kh)) * 64 + j * 16 + ln];
      }
      o[j] = wmma_bf16(pa0, vf0.v, o[j]);
      o[j] = wmma_bf16(pa1, vf1.v, o[j]);
    }
  }

  // ---- finalize: ctx = O / l (bf16), store (m, l) stats ----
#pragma unroll
  for (int e = 0; e < 8; ++e) {
    const float inv = 1.0f / lrow[e];
    const int row = t0 + e + kh * 8;
    const size_t base = ((size_t)row * BATCH + b) * EMB + hoff;
#pragma unroll
    for (int j = 0; j < 4; ++j)
      CTX[base + 16 * j + ln] = (bf16)(o[j][e] * inv);
  }
  if (ln == 0) {
#pragma unroll
    for (int e = 0; e < 8; ++e) {
      const int row = t0 + e + kh * 8;
      Mst[((size_t)b * HEADS + h) * T_LEN + row] = mrow[e];
      Lst[((size_t)b * HEADS + h) * T_LEN + row] = lrow[e];
    }
  }
}

// --- avg weights: recompute scores per head, normalize via stats ------------
// One wave per (b, 16-row t tile, 32-col s tile); loops all 16 heads (64 WMMA).
__global__ __launch_bounds__(256)
void avg_weights_kernel(const bf16* __restrict__ Q, const bf16* __restrict__ Kp,
                        const float* __restrict__ Mst, const float* __restrict__ Lst,
                        float* __restrict__ avg) {
  const int lane = threadIdx.x & 31;
  const int wv   = threadIdx.x >> 5;
  const int kh   = lane >> 4;
  const int ln   = lane & 15;

  const int gid = blockIdx.x * 8 + wv;
  const int TT = T_LEN / 16;   // 128
  const int SS = T_LEN / 32;   // 64
  const int b  = gid / (TT * SS);
  const int r  = gid % (TT * SS);
  const int t0 = (r / SS) * 16;
  const int s0 = (r % SS) * 32;
  const float scale = 0.125f;

  float w0[8] = {}, w1[8] = {};

  for (int h = 0; h < HEADS; ++h) {
    const int hoff = h * HD;
    const bf16* qrow  = Q  + ((size_t)(t0 + ln)      * BATCH + b) * EMB + hoff;
    const bf16* krow0 = Kp + ((size_t)(s0 + ln)      * BATCH + b) * EMB + hoff;
    const bf16* krow1 = Kp + ((size_t)(s0 + 16 + ln) * BATCH + b) * EMB + hoff;
    const v16bf qf0 = load_frag_contig(qrow + 0,  kh);
    const v16bf qf1 = load_frag_contig(qrow + 32, kh);
    v8f sc0 = {}, sc1 = {};
    sc0 = wmma_bf16(qf0, load_frag_contig(krow0 + 0,  kh), sc0);
    sc0 = wmma_bf16(qf1, load_frag_contig(krow0 + 32, kh), sc0);
    sc1 = wmma_bf16(qf0, load_frag_contig(krow1 + 0,  kh), sc1);
    sc1 = wmma_bf16(qf1, load_frag_contig(krow1 + 32, kh), sc1);

#pragma unroll
    for (int e = 0; e < 8; ++e) {
      const int row = t0 + e + kh * 8;
      const float m  = Mst[((size_t)b * HEADS + h) * T_LEN + row];
      const float li = 1.0f / Lst[((size_t)b * HEADS + h) * T_LEN + row];
      w0[e] += __expf(sc0[e] * scale - m) * li;
      w1[e] += __expf(sc1[e] * scale - m) * li;
    }
  }

  const float invH = 1.0f / (float)HEADS;
#pragma unroll
  for (int e = 0; e < 8; ++e) {
    const int row = t0 + e + kh * 8;
    const size_t base = ((size_t)b * T_LEN + row) * T_LEN;
    avg[base + s0 + ln]      = w0[e] * invH;
    avg[base + s0 + 16 + ln] = w1[e] * invH;
  }
}

// --- host-side orchestration ------------------------------------------------
extern "C" void kernel_launch(void* const* d_in, const int* in_sizes, int n_in,
                              void* d_out, int out_size, void* d_ws, size_t ws_size,
                              hipStream_t stream) {
  const float* query = (const float*)d_in[0];
  const float* key   = (const float*)d_in[1];
  const float* value = (const float*)d_in[2];
  const float* Wq    = (const float*)d_in[3];
  const float* bq    = (const float*)d_in[4];
  const float* Wk    = (const float*)d_in[5];
  const float* bk    = (const float*)d_in[6];
  const float* Wv    = (const float*)d_in[7];
  const float* bv    = (const float*)d_in[8];
  const float* Wo    = (const float*)d_in[9];
  const float* bo    = (const float*)d_in[10];

  const size_t XN = (size_t)T_LEN * BATCH * EMB;  // 4,194,304
  const size_t WN = (size_t)EMB * EMB;            // 1,048,576

  bf16* ws  = (bf16*)d_ws;
  bf16* xq  = ws;            bf16* xk  = xq + XN;  bf16* xv  = xk + XN;
  bf16* wq  = xv + XN;       bf16* wk  = wq + WN;
  bf16* wv_ = wk + WN;       bf16* wo  = wv_ + WN;
  bf16* qp  = wo + WN;       bf16* kp  = qp + XN;  bf16* vp  = kp + XN;
  bf16* ctx = vp + XN;
  float* Mst = (float*)(ctx + XN);
  float* Lst = Mst + (size_t)BATCH * HEADS * T_LEN;

  float* outp = (float*)d_out;      // [T,B,E]
  float* avg  = outp + XN;          // [B,T,S]

  // 1) convert inputs + weights to bf16 (vectorized x8)
  f32_to_bf16_kernel<<<512, 256, 0, stream>>>(query, xq, (int)(XN / 8));
  f32_to_bf16_kernel<<<512, 256, 0, stream>>>(key,   xk, (int)(XN / 8));
  f32_to_bf16_kernel<<<512, 256, 0, stream>>>(value, xv, (int)(XN / 8));
  f32_to_bf16_kernel<<<256, 256, 0, stream>>>(Wq, wq,  (int)(WN / 8));
  f32_to_bf16_kernel<<<256, 256, 0, stream>>>(Wk, wk,  (int)(WN / 8));
  f32_to_bf16_kernel<<<256, 256, 0, stream>>>(Wv, wv_, (int)(WN / 8));
  f32_to_bf16_kernel<<<256, 256, 0, stream>>>(Wo, wo,  (int)(WN / 8));

  // 2) projections (bf16 WMMA, f32 accumulate, bf16 out)
  dim3 gblk(EMB / 128, MROWS / 128);  // (8, 32)
  gemm_xwt_bias<false><<<gblk, 256, 0, stream>>>(xq, wq, bq, qp, MROWS, EMB, EMB);
  gemm_xwt_bias<false><<<gblk, 256, 0, stream>>>(xk, wk, bk, kp, MROWS, EMB, EMB);
  gemm_xwt_bias<false><<<gblk, 256, 0, stream>>>(xv, wv_, bv, vp, MROWS, EMB, EMB);

  // 3) flash attention -> ctx + softmax stats (one block per (b,h,128-row panel))
  flash_attn_kernel<<<BATCH * HEADS * (T_LEN / 128), 256, 0, stream>>>(
      qp, kp, vp, ctx, Mst, Lst);

  // 4) head-averaged attention weights
  avg_weights_kernel<<<(BATCH * (T_LEN / 16) * (T_LEN / 32)) / 8, 256, 0, stream>>>(
      qp, kp, Mst, Lst, avg);

  // 5) output projection (f32 out into d_out)
  gemm_xwt_bias<true><<<gblk, 256, 0, stream>>>(ctx, wo, bo, outp, MROWS, EMB, EMB);
}
